// LogicGatedSNN_1374389535032
// MI455X (gfx1250) — matrix operand order repaired and checked by reading, not compiled, not run on hardware
//
#include <hip/hip_runtime.h>

// B=16384 rows, IN=512, OUT=2048.  out = spikes [B, OUT] float32.
// d_in[0]=x [B*IN] f32, d_in[1]=w [OUT*IN] f32, d_in[2]=adaptive_threshold [OUT] f32.
// Workspace: Xh fragments (B*IN halfs = 16 MB) followed by Wh fragments (OUT*IN halfs = 2 MB).

typedef __attribute__((ext_vector_type(16))) _Float16 v16h;
typedef __attribute__((ext_vector_type(8)))  float    v8f;

#define BDIM   16384
#define INDIM  512
#define OUTDIM 2048

// ---------------------------------------------------------------------------
// Prep: per 16-row tile -> L2-normalize rows, cast to f16, emit WMMA fragments.
// Fragment layout per tile: [kstep(16)][lane(32)][j(16)] halfs, so the main
// kernel loads one contiguous v16h (32B) per lane per fragment.
//   A (isB=0), 16x32 f16 (contiguous k order, same as B): lane L -> M=L%16;
//       K = (L>=16 ? 16 : 0) + j                        (+ 32*kstep)
//   B (isB=1), 32x16 f16: lane L -> N=L%16;
//       K = (L>=16 ? 16 : 0) + j                        (+ 32*kstep)
// ---------------------------------------------------------------------------
__global__ __launch_bounds__(256) void snn_prep(const float* __restrict__ src,
                                                _Float16* __restrict__ dst,
                                                int isB) {
  __shared__ float tile[16 * INDIM];   // 32 KB
  __shared__ float rs[16];
  const int tid = threadIdx.x;
  const float* base = src + (size_t)blockIdx.x * (16 * INDIM);

  if (tid < 16) rs[tid] = 0.0f;
  __syncthreads();

  for (int i = tid; i < 16 * INDIM; i += 256) {
    float v = base[i];
    tile[i] = v;
    atomicAdd(&rs[i >> 9], v * v);     // i/512 = local row
  }
  __syncthreads();

  if (tid < 16) {
    float n = sqrtf(rs[tid]);
    rs[tid] = 1.0f / fmaxf(n, 1e-8f);  // F.normalize: x / clamp_min(||x||, eps)
  }
  __syncthreads();

  _Float16* dtile = dst + (size_t)blockIdx.x * (16 * INDIM);
  for (int idx = tid; idx < 16 * 32; idx += 256) {
    const int ks   = idx >> 5;
    const int lane = idx & 31;
    const int row  = lane & 15;        // M (A) or N (B)
    const float s  = rs[row];
#pragma unroll
    for (int j = 0; j < 16; ++j) {
      const int K = isB ? (((lane & 16) ? 16 : 0) + j)
                        : (((lane & 16) ? 16 : 0) + j);
      const float v = tile[row * INDIM + ks * 32 + K] * s;
      dtile[(size_t)idx * 16 + j] = (_Float16)v;
    }
  }
}

// ---------------------------------------------------------------------------
// Main: 16 batch rows per block, 8 waves x 256 columns each.
// 128 accumulator VGPRs/lane hold the full [16 x 2048] v_mem stripe; epilogue
// (mean, cubic, max/argmax, threshold, WTA) fused via tiny LDS reductions.
// ---------------------------------------------------------------------------
__global__ __launch_bounds__(256) void snn_main(const _Float16* __restrict__ Xh,
                                                const _Float16* __restrict__ Wh,
                                                const float* __restrict__ ath,
                                                float* __restrict__ out) {
  __shared__ float              s_sum[16];
  __shared__ unsigned long long s_max[16];
  __shared__ unsigned int       s_spk[16];

  const int tid  = threadIdx.x;
  const int wave = tid >> 5;           // 0..7
  const int lane = tid & 31;
  const int coln = lane & 15;          // N within a 16x16 tile
  const int blk  = blockIdx.x;         // M-tile index

  if (tid < 16) { s_sum[tid] = 0.0f; s_max[tid] = 0ULL; s_spk[tid] = 0u; }

  v8f zero = {};
  v8f acc[16];
#pragma unroll
  for (int i = 0; i < 16; ++i) acc[i] = zero;

  const _Float16* Ab = Xh + (size_t)blk * (16 * INDIM);
  const _Float16* Wb = Wh + (size_t)(wave * 16) * (16 * INDIM);

  for (int ks = 0; ks < 16; ++ks) {
    const v16h a = *(const v16h*)(Ab + (size_t)(ks * 32 + lane) * 16);
#pragma unroll
    for (int nt = 0; nt < 16; ++nt) {
      const v16h b = *(const v16h*)(Wb + (size_t)nt * (16 * INDIM)
                                       + (size_t)(ks * 32 + lane) * 16);
      acc[nt] = __builtin_amdgcn_wmma_f32_16x16x32_f16(
          /*neg_a=*/false, a, /*neg_b=*/false, b,
          /*c_mod=*/(short)0, acc[nt], /*reuse_a=*/false, /*reuse_b=*/false);
    }
  }
  __syncthreads();  // init of s_* visible; acc complete

  // C/D layout: VGPR r, lane L -> row = r + 8*(L>=16), col = Ntile*16 + L%16
  const int rbase = (lane & 16) ? 8 : 0;

  // 1) row sums -> mean (inhibition)
#pragma unroll
  for (int r = 0; r < 8; ++r) {
    float ps = 0.0f;
#pragma unroll
    for (int nt = 0; nt < 16; ++nt) ps += acc[nt][r];
    atomicAdd(&s_sum[rbase + r], ps);
  }
  __syncthreads();

  // 2) center + cubic expansion in place; packed max|argmax via ds_max_u64
#pragma unroll
  for (int r = 0; r < 8; ++r) {
    const float mean = s_sum[rbase + r] * (1.0f / (float)OUTDIM);
    unsigned long long best = 0ULL;
#pragma unroll
    for (int nt = 0; nt < 16; ++nt) {
      const float c  = acc[nt][r] - mean;
      const float a_ = fabsf(c);
      const float v  = copysignf((a_ + 10.0f * a_ * a_ * a_) * 50.0f, c);
      acc[nt][r] = v;
      const unsigned int bits = __float_as_uint(v);
      const unsigned int key  = (bits & 0x80000000u) ? ~bits : (bits | 0x80000000u);
      const unsigned int col  = (unsigned int)(wave * 256 + nt * 16 + coln);
      const unsigned long long p =
          ((unsigned long long)key << 32) | (unsigned long long)col;
      best = (p > best) ? p : best;
    }
    atomicMax(&s_max[rbase + r], best);
  }
  __syncthreads();

  // 3) adaptive threshold; has-spike flag per row
  float athl[16];
#pragma unroll
  for (int nt = 0; nt < 16; ++nt) athl[nt] = ath[wave * 256 + nt * 16 + coln];

#pragma unroll
  for (int r = 0; r < 8; ++r) {
    const unsigned long long pk = s_max[rbase + r];
    const unsigned int key   = (unsigned int)(pk >> 32);
    const unsigned int mbits = (key & 0x80000000u) ? (key & 0x7FFFFFFFu) : ~key;
    const float cap = __uint_as_float(mbits) * 0.25f;
    unsigned int any = 0u;
#pragma unroll
    for (int nt = 0; nt < 16; ++nt) {
      const float th = fmaxf(fminf(athl[nt], cap), 0.01f);
      any |= (acc[nt][r] >= th) ? 1u : 0u;
    }
    if (any) atomicOr(&s_spk[rbase + r], 1u);
  }
  __syncthreads();

  // 4) spikes (+ winner-take-all fallback) -> coalesced stores
#pragma unroll
  for (int r = 0; r < 8; ++r) {
    const unsigned long long pk = s_max[rbase + r];
    const unsigned int key   = (unsigned int)(pk >> 32);
    const unsigned int mcol  = (unsigned int)(pk & 0xFFFFFFFFu);
    const unsigned int mbits = (key & 0x80000000u) ? (key & 0x7FFFFFFFu) : ~key;
    const float cap = __uint_as_float(mbits) * 0.25f;
    const unsigned int has = s_spk[rbase + r];
    float* orow = out + ((size_t)blk * 16 + rbase + r) * OUTDIM;
#pragma unroll
    for (int nt = 0; nt < 16; ++nt) {
      const unsigned int col = (unsigned int)(wave * 256 + nt * 16 + coln);
      const float th = fmaxf(fminf(athl[nt], cap), 0.01f);
      float sp = (acc[nt][r] >= th) ? 1.0f : 0.0f;
      if (!has) sp = (col == mcol) ? 1.0f : 0.0f;
      orow[col] = sp;
    }
  }
}

// ---------------------------------------------------------------------------
extern "C" void kernel_launch(void* const* d_in, const int* in_sizes, int n_in,
                              void* d_out, int out_size, void* d_ws, size_t ws_size,
                              hipStream_t stream) {
  const float* x   = (const float*)d_in[0];
  const float* w   = (const float*)d_in[1];
  const float* ath = (const float*)d_in[2];
  float* out = (float*)d_out;

  _Float16* Xh = (_Float16*)d_ws;                       // 16 MB
  _Float16* Wh = Xh + (size_t)BDIM * INDIM;             // + 2 MB

  snn_prep<<<BDIM / 16,   256, 0, stream>>>(x, Xh, 0);  // A fragments
  snn_prep<<<OUTDIM / 16, 256, 0, stream>>>(w, Wh, 1);  // B fragments
  snn_main<<<BDIM / 16,   256, 0, stream>>>(Xh, Wh, ath, out);
}
